// SoftMoE_81561428951114
// MI455X (gfx1250) — matrix operand order repaired and checked by reading, not compiled, not run on hardware
//
#include <hip/hip_runtime.h>

// ---------------------------------------------------------------- constants
constexpr int kB = 8;
constexpr int kT = 4096;
constexpr int kD = 768;
constexpr int kS = 64;
constexpr int kF = 3072;

typedef __attribute__((ext_vector_type(16))) _Float16 v16h;
typedef __attribute__((ext_vector_type(8)))  _Float16 v8h;
typedef __attribute__((ext_vector_type(8)))  float    v8f;

union HF16 { v16h v; _Float16 e[16]; };

// ------------------------------------------------ CDNA5 async copy helpers
// GLOBAL_LOAD_ASYNC_TO_LDS_B128 (ISA §10.7 / §15.18.3 op 98): per-lane
// global address -> per-lane LDS address, tracked with ASYNCcnt.
__device__ inline void async_copy_b128(void* lds_dst, const void* gsrc) {
  unsigned lds = (unsigned)(uintptr_t)lds_dst;  // low 32b of generic = LDS off
  unsigned long long g = (unsigned long long)(uintptr_t)gsrc;
  asm volatile("global_load_async_to_lds_b128 %0, %1, off"
               :: "v"(lds), "v"(g) : "memory");
}
__device__ inline void async_wait0() {
  asm volatile("s_wait_asynccnt 0x0" ::: "memory");
}

// ------------------------------------------------ WMMA fragment helpers
// A fragment: 16x32 (MxK) f16, source row-major [m][k].
// m = lane&15, h = lane>>4; e[0..7] -> K=k0+8h+e, e[8..15] -> K=k0+16+8h+e-8.
__device__ inline v16h load_a_rowmajor(const _Float16* base, int m0, int k0,
                                       int ld, int lane) {
  int m = m0 + (lane & 15);
  int h = lane >> 4;
  const _Float16* p = base + (size_t)m * ld + k0 + 8 * h;
  HF16 f;
  *(v8h*)&f.e[0] = *(const v8h*)p;
  *(v8h*)&f.e[8] = *(const v8h*)(p + 16);
  return f.v;
}

// B fragment: 32x16 (KxN) f16 where the source is stored N-major ([n][k]
// row-major, i.e. B^T in memory). n = lane&15; e[0..7] -> K=k0+8h+e,
// e[8..15] -> K=k0+16+8h+e-8.
__device__ inline v16h load_b_nmajor(const _Float16* base, int n0, int k0,
                                     int ld, int lane) {
  int n = n0 + (lane & 15);
  int h = lane >> 4;
  const _Float16* p = base + (size_t)n * ld + k0 + 8 * h;
  HF16 f;
  *(v8h*)&f.e[0] = *(const v8h*)p;
  *(v8h*)&f.e[8] = *(const v8h*)(p + 16);
  return f.v;
}

// C/D fragment store: VGPR r holds (m = m0 + r + 8h, n = n0 + lane&15).
__device__ inline void store_d(float* base, int m0, int n0, int ld, int lane,
                               v8f c) {
  int n = n0 + (lane & 15);
  int h = lane >> 4;
#pragma unroll
  for (int r = 0; r < 8; ++r)
    base[(size_t)(m0 + r + 8 * h) * ld + n] = c[r];
}

// ------------------------------------------------ block reductions (256 thr)
__device__ inline float block_sum(float v, float* red) {
  int tid = threadIdx.x;
  red[tid] = v;
  __syncthreads();
  for (int off = 128; off > 0; off >>= 1) {
    if (tid < off) red[tid] += red[tid + off];
    __syncthreads();
  }
  float r = red[0];
  __syncthreads();
  return r;
}
__device__ inline float block_max(float v, float* red) {
  int tid = threadIdx.x;
  red[tid] = v;
  __syncthreads();
  for (int off = 128; off > 0; off >>= 1) {
    if (tid < off) red[tid] = fmaxf(red[tid], red[tid + off]);
    __syncthreads();
  }
  float r = red[0];
  __syncthreads();
  return r;
}

// ---------------------------------------------------------------- kernel 1
// RMSNorm + token l2-normalize. One 256-thread block per (b,t) row of 768.
__global__ void k_rmsnorm(const float* __restrict__ x,
                          const float* __restrict__ nw,
                          _Float16* __restrict__ xn_h,
                          _Float16* __restrict__ tok_h) {
  __shared__ float red[256];
  int row = blockIdx.x;  // b*T + t
  const float* xr = x + (size_t)row * kD;
  float v[3];
  float ss = 0.f;
#pragma unroll
  for (int i = 0; i < 3; ++i) {
    v[i] = xr[threadIdx.x + 256 * i];
    ss += v[i] * v[i];
  }
  float ms = block_sum(ss, red) * (1.0f / kD);
  float r = rsqrtf(ms + 1e-5f);
  float xnv[3];
  float s2 = 0.f;
#pragma unroll
  for (int i = 0; i < 3; ++i) {
    xnv[i] = v[i] * r * nw[threadIdx.x + 256 * i];
    s2 += xnv[i] * xnv[i];
  }
  float l2 = sqrtf(block_sum(s2, red));
  float inv = 1.0f / fmaxf(l2, 1e-5f);
#pragma unroll
  for (int i = 0; i < 3; ++i) {
    int d = threadIdx.x + 256 * i;
    xn_h[(size_t)row * kD + d] = (_Float16)xnv[i];
    tok_h[(size_t)row * kD + d] = (_Float16)(xnv[i] * inv);
  }
}

// ---------------------------------------------------------------- kernel 2
// Slot l2-normalize * scale. One block per slot.
__global__ void k_slots(const float* __restrict__ slots,
                        const float* __restrict__ scale,
                        _Float16* __restrict__ sl_h) {
  __shared__ float red[256];
  int s = blockIdx.x;
  const float* sr = slots + (size_t)s * kD;
  float v[3];
  float ss = 0.f;
#pragma unroll
  for (int i = 0; i < 3; ++i) {
    v[i] = sr[threadIdx.x + 256 * i];
    ss += v[i] * v[i];
  }
  float l2 = sqrtf(block_sum(ss, red));
  float inv = scale[s] / fmaxf(l2, 1e-5f);
#pragma unroll
  for (int i = 0; i < 3; ++i)
    sl_h[(size_t)s * kD + threadIdx.x + 256 * i] = (_Float16)(v[i] * inv);
}

// ---------------------------------------------------------------- kernel T
// f16 matrix transpose [R x C] -> [C x R] per batch, 64x64 LDS tiles,
// coalesced 128B rows on both the read and write side.
__global__ void k_transpose(const _Float16* __restrict__ src,
                            _Float16* __restrict__ dst, int R, int C) {
  __shared__ _Float16 tile[64][64 + 8];
  int batch = blockIdx.z;
  const _Float16* s = src + (size_t)batch * R * C;
  _Float16* d = dst + (size_t)batch * R * C;
  int r0 = blockIdx.y * 64;
  int c0 = blockIdx.x * 64;
  int tx = threadIdx.x & 63;  // column within tile
  int ty = threadIdx.x >> 6;  // 0..3
#pragma unroll
  for (int i = 0; i < 16; ++i) {
    int r = ty + i * 4;
    tile[r][tx] = s[(size_t)(r0 + r) * C + c0 + tx];
  }
  __syncthreads();
#pragma unroll
  for (int i = 0; i < 16; ++i) {
    int r = ty + i * 4;  // row of transposed tile = original column
    d[(size_t)(c0 + r) * R + r0 + tx] = tile[tx][r];
  }
}

// ---------------------------------------------------------------- kernel 3
// logits[b,s,t] = sum_d sl[s,d] * tok[b,t,d].  A = sl (row-major MxK),
// B = tok (N-major), K = D = 768 -> 24 wmma per tile. 8 waves/block.
__global__ void k_logits(const _Float16* __restrict__ sl_h,
                         const _Float16* __restrict__ tok_h,
                         float* __restrict__ logits) {
  int tile = blockIdx.x * 8 + (threadIdx.x >> 5);
  int lane = threadIdx.x & 31;
  const int MT = kS / 16, NT = kT / 16;
  int b = tile / (MT * NT);
  int r = tile % (MT * NT);
  int mt = r / NT, nt = r % NT;
  const _Float16* tokb = tok_h + (size_t)b * kT * kD;
  v8f c = {};
  for (int k0 = 0; k0 < kD; k0 += 32) {
    v16h a = load_a_rowmajor(sl_h, mt * 16, k0, kD, lane);
    v16h bb = load_b_nmajor(tokb, nt * 16, k0, kD, lane);
    c = __builtin_amdgcn_wmma_f32_16x16x32_f16(false, a, false, bb, (short)0,
                                               c, false, false);
  }
  store_d(logits + (size_t)b * kS * kT, mt * 16, nt * 16, kT, lane, c);
}

// ---------------------------------------------------------------- kernel 4
// dispatch softmax over T. One block per (b,s) row of 4096.
__global__ void k_softmax_T(const float* __restrict__ logits,
                            _Float16* __restrict__ disp) {
  __shared__ float red[256];
  int bs = blockIdx.x;
  const float* row = logits + (size_t)bs * kT;
  float v[16];
  float mx = -3.4e38f;
#pragma unroll
  for (int i = 0; i < 16; ++i) {
    v[i] = row[threadIdx.x + 256 * i];
    mx = fmaxf(mx, v[i]);
  }
  mx = block_max(mx, red);
  float sum = 0.f;
#pragma unroll
  for (int i = 0; i < 16; ++i) {
    v[i] = expf(v[i] - mx);
    sum += v[i];
  }
  float inv = 1.0f / block_sum(sum, red);
#pragma unroll
  for (int i = 0; i < 16; ++i)
    disp[(size_t)bs * kT + threadIdx.x + 256 * i] = (_Float16)(v[i] * inv);
}

// ---------------------------------------------------------------- kernel 5
// combine softmax over S (stride-T reads, coalesced across t). Writes
// transposed [b,t,s] so the combine GEMM A-fragment is contiguous.
__global__ void k_softmax_S(const float* __restrict__ logits,
                            _Float16* __restrict__ comb) {
  int idx = blockIdx.x * 256 + threadIdx.x;  // b*T + t
  int b = idx / kT;
  int t = idx % kT;
  const float* base = logits + (size_t)b * kS * kT + t;
  float v[kS];
  float mx = -3.4e38f;
#pragma unroll
  for (int s = 0; s < kS; ++s) {
    v[s] = base[(size_t)s * kT];
    mx = fmaxf(mx, v[s]);
  }
  float sum = 0.f;
#pragma unroll
  for (int s = 0; s < kS; ++s) {
    v[s] = expf(v[s] - mx);
    sum += v[s];
  }
  float inv = 1.0f / sum;
  _Float16* o = comb + (size_t)idx * kS;
#pragma unroll
  for (int s = 0; s < kS; ++s) o[s] = (_Float16)(v[s] * inv);
}

// ---------------------------------------------------------------- kernel 6
// y[b,s,d] = sum_t disp[b,s,t] * xn[b,t,d].  A = disp row-major (ld=T),
// B = xn_t[b][d][t] (N-major, ld=T) -> fully contiguous b128 clauses,
// K = 4096 -> 128 wmma per tile.
__global__ void k_ygemm(const _Float16* __restrict__ disp,
                        const _Float16* __restrict__ xn_t,
                        float* __restrict__ y) {
  int tile = blockIdx.x * 8 + (threadIdx.x >> 5);
  int lane = threadIdx.x & 31;
  const int MT = kS / 16, NT = kD / 16;
  int b = tile / (MT * NT);
  int r = tile % (MT * NT);
  int mt = r / NT, nt = r % NT;
  const _Float16* A = disp + (size_t)b * kS * kT;
  const _Float16* Bm = xn_t + (size_t)b * kD * kT;  // [d][t]
  v8f c = {};
  for (int k0 = 0; k0 < kT; k0 += 32) {
    v16h a = load_a_rowmajor(A, mt * 16, k0, kT, lane);
    v16h bb = load_b_nmajor(Bm, nt * 16, k0, kT, lane);
    c = __builtin_amdgcn_wmma_f32_16x16x32_f16(false, a, false, bb, (short)0,
                                               c, false, false);
  }
  store_d(y + (size_t)b * kS * kD, mt * 16, nt * 16, kD, lane, c);
}

// ---------------------------------------------------------------- kernel 7
// FFN1: h[b,s,f] = gelu( sum_d y[b,s,d]*fc1[s,d,f] + fc1_b[s,f] ).
// Weight-streaming f32 kernel (4 flops/byte -> HBM bound, WMMA useless at
// M=8). Activation tile (8x768 f32 = 24 KB) arrives via async LDS copies;
// weight stream is prefetched 64 rows ahead and unrolled 32-deep so the
// loads clause.
__global__ void k_ffn1(const float* __restrict__ y,
                       const float* __restrict__ fc1w,
                       const float* __restrict__ fc1b,
                       float* __restrict__ h) {
  __shared__ float yl[kB][kD];
  int s = blockIdx.x;
  int f = blockIdx.y * 256 + threadIdx.x;
#pragma unroll
  for (int i = 0; i < 6; ++i) {  // 1536 x 16B chunks / 256 threads
    int c = threadIdx.x + 256 * i;
    int ofs = c * 16;
    int bb = ofs / (kD * 4);
    int db = ofs % (kD * 4);
    async_copy_b128((char*)yl + ofs,
                    (const char*)(y + (size_t)bb * kS * kD + (size_t)s * kD) +
                        db);
  }
  async_wait0();
  __syncthreads();
  float acc[kB] = {};
  const float* w = fc1w + (size_t)s * kD * kF + f;
  for (int d0 = 0; d0 < kD; d0 += 32) {
    int pf = (d0 + 64 < kD) ? d0 + 64 : kD - 1;
    __builtin_prefetch(w + (size_t)pf * kF, 0, 0);
#pragma unroll
    for (int dd = 0; dd < 32; ++dd) {
      float wv = w[(size_t)(d0 + dd) * kF];
#pragma unroll
      for (int bb = 0; bb < kB; ++bb) acc[bb] += yl[bb][d0 + dd] * wv;
    }
  }
  float bias = fc1b[(size_t)s * kF + f];
#pragma unroll
  for (int bb = 0; bb < kB; ++bb) {
    float vv = acc[bb] + bias;
    float g = 0.5f * vv * (1.0f + erff(vv * 0.70710678118654752f));
    h[(size_t)bb * kS * kF + (size_t)s * kF + f] = g;
  }
}

// ---------------------------------------------------------------- kernel 8
// FFN2: y2[b,s,d] = sum_f h[b,s,f]*fc2[s,f,d] + fc2_b[s,d], emitted as f16.
// K tiled 4 x 768 through a 24 KB async-loaded LDS tile.
__global__ void k_ffn2(const float* __restrict__ h,
                       const float* __restrict__ fc2w,
                       const float* __restrict__ fc2b,
                       _Float16* __restrict__ y2h) {
  __shared__ float hl[kB][768];
  int s = blockIdx.x;
  int d = blockIdx.y * 256 + threadIdx.x;
  float acc[kB] = {};
  for (int c0 = 0; c0 < kF; c0 += 768) {
    __syncthreads();
#pragma unroll
    for (int i = 0; i < 6; ++i) {
      int c = threadIdx.x + 256 * i;
      int ofs = c * 16;
      int bb = ofs / (768 * 4);
      int jb = ofs % (768 * 4);
      async_copy_b128(
          (char*)hl + ofs,
          (const char*)(h + (size_t)bb * kS * kF + (size_t)s * kF + c0) + jb);
    }
    async_wait0();
    __syncthreads();
    const float* w = fc2w + ((size_t)s * kF + c0) * kD + d;
    for (int j0 = 0; j0 < 768; j0 += 32) {
      int pf = (j0 + 64 < 768) ? j0 + 64 : 767;
      __builtin_prefetch(w + (size_t)pf * kD, 0, 0);
#pragma unroll
      for (int jj = 0; jj < 32; ++jj) {
        float wv = w[(size_t)(j0 + jj) * kD];
#pragma unroll
        for (int bb = 0; bb < kB; ++bb) acc[bb] += hl[bb][j0 + jj] * wv;
      }
    }
  }
  float bias = fc2b[(size_t)s * kD + d];
#pragma unroll
  for (int bb = 0; bb < kB; ++bb)
    y2h[(size_t)bb * kS * kD + (size_t)s * kD + d] = (_Float16)(acc[bb] + bias);
}

// ---------------------------------------------------------------- kernel 9
// out[b,t,d] = sum_s y2[b,s,d] * comb[b,t,s].  A = comb row-major (ld=S=64),
// B = y2t[b][d][s] (N-major, ld=S) -> contiguous; K = 64 -> 2 wmma per tile;
// output-store bound.
__global__ void k_outgemm(const _Float16* __restrict__ comb,
                          const _Float16* __restrict__ y2t,
                          float* __restrict__ out) {
  int tile = blockIdx.x * 8 + (threadIdx.x >> 5);
  int lane = threadIdx.x & 31;
  const int MT = kT / 16, NT = kD / 16;
  int b = tile / (MT * NT);
  int r = tile % (MT * NT);
  int mt = r / NT, nt = r % NT;
  const _Float16* A = comb + (size_t)b * kT * kS;  // [t][s]
  const _Float16* Bm = y2t + (size_t)b * kD * kS;  // [d][s]
  v8f c = {};
#pragma unroll
  for (int k0 = 0; k0 < kS; k0 += 32) {
    v16h a = load_a_rowmajor(A, mt * 16, k0, kS, lane);
    v16h bb = load_b_nmajor(Bm, nt * 16, k0, kS, lane);
    c = __builtin_amdgcn_wmma_f32_16x16x32_f16(false, a, false, bb, (short)0,
                                               c, false, false);
  }
  store_d(out + (size_t)b * kT * kD, mt * 16, nt * 16, kD, lane, c);
}

// ---------------------------------------------------------------- launch
static inline size_t align256(size_t x) { return (x + 255) & ~(size_t)255; }

extern "C" void kernel_launch(void* const* d_in, const int* in_sizes, int n_in,
                              void* d_out, int out_size, void* d_ws,
                              size_t ws_size, hipStream_t stream) {
  (void)in_sizes; (void)n_in; (void)out_size; (void)ws_size;
  const float* x      = (const float*)d_in[0];
  const float* slots  = (const float*)d_in[1];
  const float* scale  = (const float*)d_in[2];
  const float* fc1w   = (const float*)d_in[3];
  const float* fc1b   = (const float*)d_in[4];
  const float* fc2w   = (const float*)d_in[5];
  const float* fc2b   = (const float*)d_in[6];
  const float* norm_w = (const float*)d_in[7];
  float* out = (float*)d_out;

  char* ws = (char*)d_ws;
  size_t cur = 0;
  auto alloc = [&](size_t bytes) -> void* {
    void* p = ws + cur;
    cur = align256(cur + bytes);
    return p;
  };
  _Float16* xn_h  = (_Float16*)alloc((size_t)kB * kT * kD * 2);
  _Float16* xn_t  = (_Float16*)alloc((size_t)kB * kD * kT * 2);
  _Float16* tok_h = (_Float16*)alloc((size_t)kB * kT * kD * 2);
  _Float16* sl_h  = (_Float16*)alloc((size_t)kS * kD * 2);
  float*    logit = (float*)   alloc((size_t)kB * kS * kT * 4);
  _Float16* disp  = (_Float16*)alloc((size_t)kB * kS * kT * 2);
  _Float16* comb  = (_Float16*)alloc((size_t)kB * kT * kS * 2);
  float*    yv    = (float*)   alloc((size_t)kB * kS * kD * 4);
  float*    hv    = (float*)   alloc((size_t)kB * kS * kF * 4);
  _Float16* y2h   = (_Float16*)alloc((size_t)kB * kS * kD * 2);
  _Float16* y2t   = (_Float16*)alloc((size_t)kB * kD * kS * 2);

  k_rmsnorm<<<kB * kT, 256, 0, stream>>>(x, norm_w, xn_h, tok_h);
  k_transpose<<<dim3(kD / 64, kT / 64, kB), 256, 0, stream>>>(xn_h, xn_t, kT,
                                                              kD);
  k_slots<<<kS, 256, 0, stream>>>(slots, scale, sl_h);
  k_logits<<<(kB * (kS / 16) * (kT / 16)) / 8, 256, 0, stream>>>(sl_h, tok_h,
                                                                 logit);
  k_softmax_T<<<kB * kS, 256, 0, stream>>>(logit, disp);
  k_softmax_S<<<(kB * kT) / 256, 256, 0, stream>>>(logit, comb);
  k_ygemm<<<(kB * (kS / 16) * (kD / 16)) / 8, 256, 0, stream>>>(disp, xn_t,
                                                                yv);
  k_ffn1<<<dim3(kS, kF / 256), 256, 0, stream>>>(yv, fc1w, fc1b, hv);
  k_ffn2<<<dim3(kS, kD / 256), 256, 0, stream>>>(hv, fc2w, fc2b, y2h);
  k_transpose<<<dim3(kD / 64, kS / 64, kB), 256, 0, stream>>>(y2h, y2t, kS,
                                                              kD);
  k_outgemm<<<(kB * (kT / 16) * (kD / 16)) / 8, 256, 0, stream>>>(comb, y2t,
                                                                  out);
}
